// WaveAudio1D_60885456388560
// MI455X (gfx1250) — matrix operand order repaired and checked by reading, not compiled, not run on hardware
//
#include <hip/hip_runtime.h>
#include <cstddef>

// ---------------------------------------------------------------------------
// Model dims
// ---------------------------------------------------------------------------
#define Bsz   128
#define Lsz   64000
#define Psz   400
#define PATCH 160
#define Esz   256
#define NLs   4
#define NCs   35
#define Mtok  (Bsz * Psz)        // 51200 rows of tokens
#define FF    (4 * Esz)          // 1024

// ---------------------------------------------------------------------------
// gfx1250 async LDS DMA path (GLOBAL_LOAD_ASYNC_TO_LDS_B128, ASYNCcnt),
// guarded so missing builtins fall back to VGPR staging + ds_store.
// Probe-learned signature: (int4vec AS1* src, int4vec AS3* dst, Ii, Ii)
// ---------------------------------------------------------------------------
#if defined(__AMDGCN__) && \
    __has_builtin(__builtin_amdgcn_global_load_async_to_lds_b128) && \
    __has_builtin(__builtin_amdgcn_s_wait_asynccnt)
#define HAVE_ASYNC_LDS 1
#else
#define HAVE_ASYNC_LDS 0
#endif

typedef int v4i_async __attribute__((vector_size(16)));

// ---------------------------------------------------------------------------
// WMMA vector types (gfx1250, wave32)
// ---------------------------------------------------------------------------
typedef __bf16 v16bf __attribute__((ext_vector_type(16)));
typedef __bf16 v8bf  __attribute__((ext_vector_type(8)));
typedef float  v8f   __attribute__((ext_vector_type(8)));

union BF16Frag { v16bf v; v8bf h[2]; };

// ---------------------------------------------------------------------------
// f32 -> bf16 conversion / zero
// ---------------------------------------------------------------------------
__global__ __launch_bounds__(256) void cvt_f2bf(const float* __restrict__ in,
                                                __bf16* __restrict__ out, int n) {
  int i = blockIdx.x * 256 + threadIdx.x;
  if (i < n) out[i] = (__bf16)in[i];
}

__global__ __launch_bounds__(256) void zero_f32(float* __restrict__ p, int n) {
  int i = blockIdx.x * 256 + threadIdx.x;
  if (i < n) p[i] = 0.f;
}

// ---------------------------------------------------------------------------
// Tiled WMMA GEMM:  out[M,N] = A[M,K] * Bt[N,K]^T  (bf16 in, f32 acc)
// Block tile 128x128, BK=32, 8 waves; per-wave tile 32x64 (2x4 16x16 frags).
// Double-buffered LDS, one barrier per K-step; tiles DMA'd into LDS with
// global_load_async_to_lds_b128 when the toolchain exposes it.
//   MODE 0: out_f32 = acc + bias[n] + pos[(m%P)*N + n]       (patch embed)
//   MODE 1: ssum[b*N+n] += (acc+bias)^2  summed over m       (z column norms)
//   MODE 2: out_bf16 = gelu(acc + bias)                      (FFN up)
//   MODE 3: out_f32[m*N+n] += acc + bias                     (FFN down resid)
// Requires: M%128==0, N%128==0, K%32==0  (true for all uses here)
// ---------------------------------------------------------------------------
#define BM 128
#define BN 128
#define BK 32
#define LDS_STRIDE 40   // 32 + 8 pad (80B rows, 16B aligned)

template <int MODE>
__global__ __launch_bounds__(256)
void gemm_wmma_bf16(const __bf16* __restrict__ A,
                    const __bf16* __restrict__ Bt,
                    const float*  __restrict__ bias,
                    float*        __restrict__ out_f32,
                    __bf16*       __restrict__ out_bf16,
                    const float*  __restrict__ pos,
                    float*        __restrict__ ssum,
                    int M, int N, int K) {
  __shared__ __align__(16) __bf16 sA[2][BM * LDS_STRIDE];
  __shared__ __align__(16) __bf16 sB[2][BN * LDS_STRIDE];

  const int tid  = threadIdx.x;
  const int wave = tid >> 5;
  const int lane = tid & 31;
  const int wm   = wave & 3;        // 4 waves along M (32 rows each)
  const int wn   = wave >> 2;       // 2 waves along N (64 cols each)
  const int m0   = blockIdx.y * BM;
  const int n0   = blockIdx.x * BN;
  const int r    = lane & 15;
  const int hi   = lane >> 4;

  v8f acc[2][4];
#pragma unroll
  for (int i = 0; i < 2; ++i)
#pragma unroll
    for (int j = 0; j < 4; ++j)
#pragma unroll
      for (int e = 0; e < 8; ++e) acc[i][j][e] = 0.f;

  // staging chunk geometry: 128 rows x 4 x 16B chunks = 512 chunks, 2/thread
#if HAVE_ASYNC_LDS
  auto async_tile = [&](int k0, int buf) {
#pragma unroll
    for (int it = 0; it < 2; ++it) {
      const int c   = tid + it * 256;
      const int row = c >> 2;
      const int c8  = (c & 3) * 8;
      __builtin_amdgcn_global_load_async_to_lds_b128(
          (__attribute__((address_space(1))) v4i_async*)(A + (size_t)(m0 + row) * K + k0 + c8),
          (__attribute__((address_space(3))) v4i_async*)(&sA[buf][row * LDS_STRIDE + c8]),
          0, 0);
      __builtin_amdgcn_global_load_async_to_lds_b128(
          (__attribute__((address_space(1))) v4i_async*)(Bt + (size_t)(n0 + row) * K + k0 + c8),
          (__attribute__((address_space(3))) v4i_async*)(&sB[buf][row * LDS_STRIDE + c8]),
          0, 0);
    }
  };
#else
  v8bf ra[2], rb[2];
  auto load_tile = [&](int k0) {
#pragma unroll
    for (int it = 0; it < 2; ++it) {
      const int c   = tid + it * 256;
      const int row = c >> 2;
      const int c8  = (c & 3) * 8;
      ra[it] = *(const v8bf*)(A  + (size_t)(m0 + row) * K + k0 + c8);
      rb[it] = *(const v8bf*)(Bt + (size_t)(n0 + row) * K + k0 + c8);
    }
  };
  auto store_tile = [&](int buf) {
#pragma unroll
    for (int it = 0; it < 2; ++it) {
      const int c   = tid + it * 256;
      const int row = c >> 2;
      const int c8  = (c & 3) * 8;
      *(v8bf*)(&sA[buf][row * LDS_STRIDE + c8]) = ra[it];
      *(v8bf*)(&sB[buf][row * LDS_STRIDE + c8]) = rb[it];
    }
  };
#endif

#if HAVE_ASYNC_LDS
  async_tile(0, 0);
  __builtin_amdgcn_s_wait_asynccnt(0);
  __syncthreads();
#else
  load_tile(0);
  store_tile(0);
  __syncthreads();
#endif

  const int nk = K / BK;
  for (int ks = 0; ks < nk; ++ks) {
    const int buf = ks & 1;

    // launch DMA / fetch of the next tile while this tile computes
#if HAVE_ASYNC_LDS
    if (ks + 1 < nk) async_tile((ks + 1) * BK, buf ^ 1);
#else
    if (ks + 1 < nk) load_tile((ks + 1) * BK);
#endif

    // L2 prefetch two tiles ahead (global_prefetch_b8)
    if (ks + 2 < nk) {
#pragma unroll
      for (int it = 0; it < 2; ++it) {
        const int c   = tid + it * 256;
        const int row = c >> 2;
        const int c8  = (c & 3) * 8;
        __builtin_prefetch(A  + (size_t)(m0 + row) * K + (ks + 2) * BK + c8, 0, 3);
        __builtin_prefetch(Bt + (size_t)(n0 + row) * K + (ks + 2) * BK + c8, 0, 3);
      }
    }

    // A frag (16x32): lane r = row M=r; elems 0-7 = K hi*8.., elems 8-15 = K 16+hi*8..
    BF16Frag aF[2], bF[4];
#pragma unroll
    for (int i = 0; i < 2; ++i) {
      const __bf16* Ar = &sA[buf][(wm * 32 + i * 16 + r) * LDS_STRIDE];
      aF[i].h[0] = *(const v8bf*)(Ar + hi * 8);
      aF[i].h[1] = *(const v8bf*)(Ar + 16 + hi * 8);
    }
    // B frag (32x16): lane r = column N=r; elems 0-7 = K hi*8.., elems 8-15 = K 16+hi*8..
#pragma unroll
    for (int j = 0; j < 4; ++j) {
      const __bf16* Br = &sB[buf][(wn * 64 + j * 16 + r) * LDS_STRIDE];
      bF[j].h[0] = *(const v8bf*)(Br + hi * 8);
      bF[j].h[1] = *(const v8bf*)(Br + 16 + hi * 8);
    }

#pragma unroll
    for (int i = 0; i < 2; ++i)
#pragma unroll
      for (int j = 0; j < 4; ++j)
        acc[i][j] = __builtin_amdgcn_wmma_f32_16x16x32_bf16(
            false, aF[i].v, false, bF[j].v, (short)0, acc[i][j], false, false);

    // make the next tile resident in LDS, then one barrier
#if HAVE_ASYNC_LDS
    if (ks + 1 < nk) __builtin_amdgcn_s_wait_asynccnt(0);
#else
    if (ks + 1 < nk) store_tile(buf ^ 1);
#endif
    __syncthreads();
  }

  // epilogue: C/D layout -> element e of acc = (M = hi*8+e, N = r) within frag
#pragma unroll
  for (int i = 0; i < 2; ++i) {
    const int fm = m0 + wm * 32 + i * 16 + hi * 8;   // first of 8 rows (same batch)
#pragma unroll
    for (int j = 0; j < 4; ++j) {
      const int gn = n0 + wn * 64 + j * 16 + r;
      const float bv = bias[gn];
      if constexpr (MODE == 1) {
        float ss = 0.f;
#pragma unroll
        for (int e = 0; e < 8; ++e) {
          float v = acc[i][j][e] + bv;
          ss += v * v;
        }
        atomicAdd(&ssum[(fm / Psz) * N + gn], ss);
      } else {
#pragma unroll
        for (int e = 0; e < 8; ++e) {
          const int gm = fm + e;
          float v = acc[i][j][e] + bv;
          if constexpr (MODE == 0) {
            out_f32[(size_t)gm * N + gn] = v + pos[(gm % Psz) * N + gn];
          } else if constexpr (MODE == 2) {
            float g = 0.5f * v * (1.f + erff(v * 0.70710678118654752f));
            out_bf16[(size_t)gm * N + gn] = (__bf16)g;
          } else {  // MODE 3
            out_f32[(size_t)gm * N + gn] += v;
          }
        }
      }
    }
  }
}

// ---------------------------------------------------------------------------
// LayerNorm over E=256 per row: one wave32 per row, 8 rows per block.
// f32 in -> bf16 out (feeds WMMA GEMMs).
// ---------------------------------------------------------------------------
__global__ __launch_bounds__(256)
void layernorm_rows(const float* __restrict__ x, __bf16* __restrict__ out,
                    const float* __restrict__ g, const float* __restrict__ b,
                    int Mrows) {
  const int wave = threadIdx.x >> 5;
  const int lane = threadIdx.x & 31;
  const int row  = blockIdx.x * 8 + wave;
  if (row >= Mrows) return;
  const float* xr = x + (size_t)row * Esz;
  float v[8];
  *(float4*)&v[0] = *(const float4*)(xr + lane * 8);
  *(float4*)&v[4] = *(const float4*)(xr + lane * 8 + 4);
  float s = 0.f, sq = 0.f;
#pragma unroll
  for (int e = 0; e < 8; ++e) { s += v[e]; sq += v[e] * v[e]; }
#pragma unroll
  for (int m = 16; m >= 1; m >>= 1) {
    s  += __shfl_xor(s, m, 32);
    sq += __shfl_xor(sq, m, 32);
  }
  const float mu   = s * (1.f / Esz);
  const float var  = sq * (1.f / Esz) - mu * mu;
  const float rstd = rsqrtf(var + 1e-5f);
#pragma unroll
  for (int e = 0; e < 8; ++e) {
    const int eg = lane * 8 + e;
    out[(size_t)row * Esz + eg] = (__bf16)((v[e] - mu) * rstd * g[eg] + b[eg]);
  }
}

// ---------------------------------------------------------------------------
// wave residual: x[b,p,e] += sqrt(ss1[b,e]+eps)*sqrt(ss2[b,e]+eps)
// (|g1*g2*exp(i phi)| == g1*g2 : phases cancel exactly)
// ---------------------------------------------------------------------------
__global__ __launch_bounds__(256)
void wave_residual(float* __restrict__ x, const float* __restrict__ s1,
                   const float* __restrict__ s2) {
  const size_t i = (size_t)blockIdx.x * 256 + threadIdx.x;
  const int e = (int)(i & (Esz - 1));
  const int b = (int)(i / ((size_t)Psz * Esz));
  const float g1 = sqrtf(s1[b * Esz + e] + 1e-8f);
  const float g2 = sqrtf(s2[b * Esz + e] + 1e-8f);
  x[i] += g1 * g2;
}

// ---------------------------------------------------------------------------
// mean over P then final LayerNorm -> nbuf[B,E] f32
// ---------------------------------------------------------------------------
__global__ __launch_bounds__(256)
void mean_ln(const float* __restrict__ x, const float* __restrict__ g,
             const float* __restrict__ b, float* __restrict__ nbuf) {
  const int bi = blockIdx.x, e = threadIdx.x;
  const float* xr = x + (size_t)bi * Psz * Esz + e;
  float s = 0.f;
  for (int p = 0; p < Psz; ++p) s += xr[(size_t)p * Esz];
  const float m = s * (1.f / Psz);
  __shared__ float red[Esz], red2[Esz];
  red[e] = m; red2[e] = m * m;
  __syncthreads();
  for (int st = 128; st > 0; st >>= 1) {
    if (e < st) { red[e] += red[e + st]; red2[e] += red2[e + st]; }
    __syncthreads();
  }
  const float mu   = red[0] * (1.f / Esz);
  const float var  = red2[0] * (1.f / Esz) - mu * mu;
  const float rstd = rsqrtf(var + 1e-5f);
  nbuf[bi * Esz + e] = (m - mu) * rstd * g[e] + b[e];
}

// ---------------------------------------------------------------------------
// classifier: out[b,c] = nbuf[b,:] . cls_w[c,:] + cls_b[c]
// ---------------------------------------------------------------------------
__global__ __launch_bounds__(64)
void logits_kernel(const float* __restrict__ nbuf, const float* __restrict__ cw,
                   const float* __restrict__ cb, float* __restrict__ out) {
  const int bi = blockIdx.x, c = threadIdx.x;
  if (c >= NCs) return;
  float acc = cb[c];
  const float* xr = nbuf + (size_t)bi * Esz;
  const float* wr = cw + (size_t)c * Esz;
  for (int e = 0; e < Esz; ++e) acc += xr[e] * wr[e];
  out[bi * NCs + c] = acc;
}

// ---------------------------------------------------------------------------
// host launcher
// ---------------------------------------------------------------------------
extern "C" void kernel_launch(void* const* d_in, const int* in_sizes, int n_in,
                              void* d_out, int out_size, void* d_ws, size_t ws_size,
                              hipStream_t stream) {
  const float* x       = (const float*)d_in[0];
  const float* patch_w = (const float*)d_in[1];
  const float* patch_b = (const float*)d_in[2];
  const float* pos     = (const float*)d_in[3];
  const float* w1      = (const float*)d_in[4];
  const float* bw1     = (const float*)d_in[5];
  const float* w2      = (const float*)d_in[6];
  const float* bw2     = (const float*)d_in[7];
  const float* ln1_g   = (const float*)d_in[8];
  const float* ln1_b   = (const float*)d_in[9];
  const float* fw1     = (const float*)d_in[10];
  const float* fb1     = (const float*)d_in[11];
  const float* fw2     = (const float*)d_in[12];
  const float* fb2     = (const float*)d_in[13];
  const float* ln2_g   = (const float*)d_in[14];
  const float* ln2_b   = (const float*)d_in[15];
  const float* norm_g  = (const float*)d_in[16];
  const float* norm_b  = (const float*)d_in[17];
  const float* cls_w   = (const float*)d_in[18];
  const float* cls_b   = (const float*)d_in[19];
  float* out = (float*)d_out;

  char* ws = (char*)d_ws;
  size_t off = 0;
  auto alloc = [&](size_t bytes) -> char* {
    char* p = ws + off;
    off += (bytes + 255) & ~(size_t)255;
    return p;
  };
  float*  xbuf = (float*) alloc((size_t)Mtok * Esz * 4);   // activations f32
  __bf16* hn   = (__bf16*)alloc((size_t)Mtok * Esz * 2);   // normed acts bf16
  __bf16* tbuf = (__bf16*)alloc((size_t)Mtok * FF * 2);    // FFN hidden bf16
  __bf16* xb   = (__bf16*)alloc((size_t)Bsz * Lsz * 2);    // raw input bf16
  __bf16* pwb  = (__bf16*)alloc((size_t)Esz * PATCH * 2);
  __bf16* w1b  = (__bf16*)alloc((size_t)NLs * Esz * Esz * 2);
  __bf16* w2b  = (__bf16*)alloc((size_t)NLs * Esz * Esz * 2);
  __bf16* fw1b = (__bf16*)alloc((size_t)NLs * FF * Esz * 2);
  __bf16* fw2b = (__bf16*)alloc((size_t)NLs * Esz * FF * 2);
  float*  ssum = (float*) alloc((size_t)2 * Bsz * Esz * 4);
  float*  nbuf = (float*) alloc((size_t)Bsz * Esz * 4);

  auto cvt = [&](const float* src, __bf16* dst, int n) {
    cvt_f2bf<<<(n + 255) / 256, 256, 0, stream>>>(src, dst, n);
  };
  cvt(x,       xb,   Bsz * Lsz);
  cvt(patch_w, pwb,  Esz * PATCH);
  cvt(w1,      w1b,  NLs * Esz * Esz);
  cvt(w2,      w2b,  NLs * Esz * Esz);
  cvt(fw1,     fw1b, NLs * FF * Esz);
  cvt(fw2,     fw2b, NLs * Esz * FF);

  // patch embed: [Mtok,160] x [256,160]^T -> xbuf (+patch_b +pos)
  gemm_wmma_bf16<0><<<dim3(Esz / BN, Mtok / BM), 256, 0, stream>>>(
      xb, pwb, patch_b, xbuf, nullptr, pos, nullptr, Mtok, Esz, PATCH);

  for (int l = 0; l < NLs; ++l) {
    layernorm_rows<<<Mtok / 8, 256, 0, stream>>>(xbuf, hn, ln1_g + l * Esz,
                                                 ln1_b + l * Esz, Mtok);
    zero_f32<<<(2 * Bsz * Esz) / 256, 256, 0, stream>>>(ssum, 2 * Bsz * Esz);
    gemm_wmma_bf16<1><<<dim3(Esz / BN, Mtok / BM), 256, 0, stream>>>(
        hn, w1b + (size_t)l * Esz * Esz, bw1 + l * Esz, nullptr, nullptr,
        nullptr, ssum, Mtok, Esz, Esz);
    gemm_wmma_bf16<1><<<dim3(Esz / BN, Mtok / BM), 256, 0, stream>>>(
        hn, w2b + (size_t)l * Esz * Esz, bw2 + l * Esz, nullptr, nullptr,
        nullptr, ssum + Bsz * Esz, Mtok, Esz, Esz);
    wave_residual<<<(size_t)Mtok * Esz / 256, 256, 0, stream>>>(
        xbuf, ssum, ssum + Bsz * Esz);
    layernorm_rows<<<Mtok / 8, 256, 0, stream>>>(xbuf, hn, ln2_g + l * Esz,
                                                 ln2_b + l * Esz, Mtok);
    gemm_wmma_bf16<2><<<dim3(FF / BN, Mtok / BM), 256, 0, stream>>>(
        hn, fw1b + (size_t)l * FF * Esz, fb1 + l * FF, nullptr, tbuf,
        nullptr, nullptr, Mtok, FF, Esz);
    gemm_wmma_bf16<3><<<dim3(Esz / BN, Mtok / BM), 256, 0, stream>>>(
        tbuf, fw2b + (size_t)l * Esz * FF, fb2 + l * Esz, xbuf, nullptr,
        nullptr, nullptr, Mtok, Esz, FF);
  }

  mean_ln<<<Bsz, Esz, 0, stream>>>(xbuf, norm_g, norm_b, nbuf);
  logits_kernel<<<Bsz, 64, 0, stream>>>(nbuf, cls_w, cls_b, out);
}
